// HyperGraphAttentionLayerSparse_63118839382181
// MI455X (gfx1250) — matrix operand order repaired and checked by reading, not compiled, not run on hardware
//
#include <hip/hip_runtime.h>
#include <hip/hip_bf16.h>

// ---------------------------------------------------------------------------
// HyperGraphAttentionLayerSparse on MI455X (gfx1250, wave32)
//
// N=30000 nodes, E=4000 hyperedges, DEG=32 pairs/node (nnz=960000), D=256.
// Working set (~90MB) fits in the 192MB L2 -> L2-bandwidth bound after the
// first pass. attn.T@X_proj (61 GFLOP dense) is decomposed into a rank-1
// "softmax background" + sparse 245M-FMA scatter. Projection GEMM runs on
// v_wmma_f32_16x16x32_bf16; node-side gathers stage rows into LDS with
// global_load_async_to_lds_b128 (ASYNCcnt) when the toolchain provides it.
// ---------------------------------------------------------------------------

#define NN    30000      // nodes
#define NE    4000       // hyperedges
#define DEG   32         // pairs per node (== wave32 lanes)
#define D     256        // feature dim
#define NNZ   (NN * DEG)
#define ALPHA 0.2f
#define PPB   8          // pairs per block in scatter kernels

typedef __attribute__((ext_vector_type(16))) __bf16 v16bf;
typedef __attribute__((ext_vector_type(8)))  float  v8f;

#if defined(__has_builtin)
#if __has_builtin(__builtin_amdgcn_global_load_async_to_lds_b128) && \
    __has_builtin(__builtin_amdgcn_s_wait_asynccnt)
#define HAVE_ASYNC_LDS 1
#endif
#endif

#ifdef HAVE_ASYNC_LDS
// Builtin signature (from hipcc diagnostic): params are pointers to 4xint
// vectors; src in global (AS1), dst in LDS (AS3), then imm offset + cpol.
typedef int v4i_t __attribute__((vector_size(16)));
typedef v4i_t __attribute__((address_space(1))) gv4i_t;
typedef v4i_t __attribute__((address_space(3))) sv4i_t;

__device__ __forceinline__ void async_b128_to_lds(const float* gsrc, float* ldst) {
  __builtin_amdgcn_global_load_async_to_lds_b128(
      (gv4i_t*)gsrc, (sv4i_t*)ldst, 0, 0);
}
#endif

// ----------------------------- device helpers ------------------------------

__device__ __forceinline__ unsigned short f2bf(float f) {
  unsigned u = __float_as_uint(f);
  u += 0x7fffu + ((u >> 16) & 1u);      // round to nearest even
  return (unsigned short)(u >> 16);
}

__device__ __forceinline__ float wave_sum(float v) {
  #pragma unroll
  for (int m = 16; m > 0; m >>= 1) v += __shfl_xor(v, m, 32);
  return v;
}
__device__ __forceinline__ float wave_max(float v) {
  #pragma unroll
  for (int m = 16; m > 0; m >>= 1) v = fmaxf(v, __shfl_xor(v, m, 32));
  return v;
}

// ------------------------------- kernels -----------------------------------

__global__ __launch_bounds__(256) void k_zero(float* p, int count) {
  int i = blockIdx.x * blockDim.x + threadIdx.x;
  if (i < count) p[i] = 0.0f;
}

__global__ __launch_bounds__(256) void k_tobf(const float* __restrict__ x,
                                              unsigned short* __restrict__ xb,
                                              int count) {
  int i = blockIdx.x * blockDim.x + threadIdx.x;
  if (i < count) xb[i] = f2bf(x[i]);
}

// Pre-swizzle weight[K=256][N=256] into per-(ktile,ntile,lane) 16-element
// B-fragments so the WMMA B operand is two contiguous b128 loads per lane.
// Fragment layout (16-bit B, 32x16): lanes 0-15 hold K=kt*32+{0..7,16..23} of col
// N=lane; lanes 16-31 hold K=kt*32+{8..15,24..31} of col N=lane-16.
__global__ __launch_bounds__(256) void k_swz(const float* __restrict__ w,
                                             unsigned short* __restrict__ wsw) {
  int tid = blockIdx.x * blockDim.x + threadIdx.x;   // (kt*16 + nt)*32 + lane
  if (tid >= (D / 32) * (D / 16) * 32) return;
  int lane = tid & 31;
  int nt   = (tid >> 5) & 15;
  int kt   = tid >> 9;
  int col  = nt * 16 + (lane & 15);
  int kb   = kt * 32 + ((lane >> 4) << 3);
  #pragma unroll
  for (int i = 0; i < 16; ++i)
    wsw[tid * 16 + i] = f2bf(w[(kb + (i & 7) + ((i >> 3) << 4)) * D + col]);
}

__global__ __launch_bounds__(256) void k_count(const int* __restrict__ pn,
                                               const int* __restrict__ pe,
                                               float* __restrict__ dv,
                                               float* __restrict__ de) {
  int p = blockIdx.x * blockDim.x + threadIdx.x;
  if (p < NNZ) {
    atomicAdd(&dv[pn[p]], 1.0f);
    atomicAdd(&de[pe[p]], 1.0f);
  }
}

__global__ __launch_bounds__(256) void k_invdeg(const float* __restrict__ dv,
                                                const float* __restrict__ de,
                                                float* __restrict__ dvinv,
                                                float* __restrict__ deinv) {
  int i = blockIdx.x * blockDim.x + threadIdx.x;
  if (i < NN) dvinv[i] = rsqrtf(dv[i]);
  else if (i < NN + NE) deinv[i - NN] = 1.0f / de[i - NN];
}

// X_proj = x @ W via WMMA bf16. One wave computes a 16x64 output tile
// (4 accumulators), looping K in steps of 32 (8 WMMA chains per acc).
__global__ __launch_bounds__(256) void k_gemm(const unsigned short* __restrict__ xb,
                                              const unsigned short* __restrict__ wsw,
                                              float* __restrict__ Xp) {
  const int wave = blockIdx.x * (blockDim.x >> 5) + (threadIdx.x >> 5);
  const int lane = threadIdx.x & 31;
  const int nT64 = D / 64;                          // 4 col-strips
  if (wave >= (NN / 16) * nT64) return;             // wave-uniform guard
  const int m0 = (wave / nT64) * 16;
  const int n0 = (wave % nT64) * 64;

  v8f acc[4] = {};
  // A fragment addressing (16-bit A, 16x32): lanes 0-15 need row M=lane,
  // K {0..7,16..23}; lanes 16-31 need row M=lane-16, K {8..15,24..31}.
  const unsigned short* arow = xb + (m0 + (lane & 15)) * D;
  const int koffA = (lane >> 4) * 8;

  for (int kt = 0; kt < D / 32; ++kt) {
    union { uint4 u[2]; v16bf v; } A;
    const int kb = kt * 32 + koffA;
    A.u[0] = *(const uint4*)(arow + kb);
    A.u[1] = *(const uint4*)(arow + kb + 16);
    #pragma unroll
    for (int j = 0; j < 4; ++j) {
      union { uint4 u[2]; v16bf v; } B;
      const unsigned short* bp =
          wsw + ((kt * (D / 16) + (n0 >> 4) + j) * 32 + lane) * 16;
      B.u[0] = *(const uint4*)(bp);
      B.u[1] = *(const uint4*)(bp + 8);
      acc[j] = __builtin_amdgcn_wmma_f32_16x16x32_bf16(
          false, A.v, false, B.v, (short)0, acc[j], false, false);
    }
  }
  // C layout: VGPR r <-> row m0 + r + 8*(lane>>4), col n0 + j*16 + (lane&15)
  const int crow = (lane >> 4) * 8;
  const int ccol = lane & 15;
  #pragma unroll
  for (int j = 0; j < 4; ++j) {
    float* cp = Xp + (m0 + crow) * D + n0 + j * 16 + ccol;
    #pragma unroll
    for (int r = 0; r < 8; ++r) cp[r * D] = acc[j][r];
  }
}

// E_feat[e] += X_proj[n] * dv^-1/2[n].  8 pairs per block: indices become
// wave-uniform SGPR loads, 8 independent load->atomic chains pipeline
// against L2 latency (E2 is 4MB -> L2-resident).
__global__ __launch_bounds__(256) void k_scatter_E(const int* __restrict__ pn,
                                                   const int* __restrict__ pe,
                                                   const float* __restrict__ Xp,
                                                   const float* __restrict__ dvinv,
                                                   float* __restrict__ E2) {
  const int base = blockIdx.x * PPB;
  const int c = threadIdx.x;
  int ns[PPB], es[PPB];
  #pragma unroll
  for (int q = 0; q < PPB; ++q) { ns[q] = pn[base + q]; es[q] = pe[base + q]; }
  #pragma unroll
  for (int q = 0; q < PPB; ++q) {
    float v = Xp[(size_t)ns[q] * D + c] * dvinv[ns[q]];
    atomicAdd(&E2[(size_t)es[q] * D + c], v);
  }
}

// Y_hat[n] = (sum_j E_feat[e_j]/de[e_j]) * dv^-1/2[n] + X_proj[n]
// Pairs of node n are n*DEG..n*DEG+DEG-1 (reference's row-grouped layout).
// The 32 gathered rows are staged to LDS with async copies when available.
__global__ __launch_bounds__(256) void k_yhat(const int* __restrict__ pe,
                                              const float* __restrict__ E2,
                                              const float* __restrict__ deinv,
                                              const float* __restrict__ dvinv,
                                              const float* __restrict__ Xp,
                                              float* __restrict__ Yh) {
  int n = blockIdx.x, c = threadIdx.x;
  __shared__ int   se[DEG];
  __shared__ float sd[DEG];
  if (threadIdx.x < DEG) {
    int e = pe[n * DEG + threadIdx.x];
    se[threadIdx.x] = e;
    sd[threadIdx.x] = deinv[e];
  }
  __syncthreads();
  float acc = 0.0f;
#ifdef HAVE_ASYNC_LDS
  __shared__ float tile[DEG * D];                 // 32KB staged gather
  #pragma unroll
  for (int it = 0; it < (DEG * D / 4) / 256; ++it) {   // 8 iters of b128/lane
    int chunk = it * 256 + threadIdx.x;
    int row = chunk >> 6;                         // 64 x b128 per row
    int col = (chunk & 63) << 2;
    async_b128_to_lds(E2 + (size_t)se[row] * D + col, &tile[row * D + col]);
  }
  __builtin_amdgcn_s_wait_asynccnt(0);
  __syncthreads();
  #pragma unroll 8
  for (int j = 0; j < DEG; ++j) acc += tile[j * D + c] * sd[j];
#else
  #pragma unroll 8
  for (int j = 0; j < DEG; ++j) acc += E2[(size_t)se[j] * D + c] * sd[j];
#endif
  Yh[n * D + c] = acc * dvinv[n] + Xp[n * D + c];
}

// s1[n] = Yh[n]·a[:D], s2[n] = Yh[n]·a[D:]  (one wave per node)
__global__ __launch_bounds__(256) void k_scores(const float* __restrict__ Yh,
                                                const float* __restrict__ a,
                                                float* __restrict__ s1,
                                                float* __restrict__ s2) {
  int n = blockIdx.x * 8 + (threadIdx.x >> 5);
  int lane = threadIdx.x & 31;
  if (n >= NN) return;
  const float* y = Yh + n * D;
  float v1 = 0.0f, v2 = 0.0f;
  #pragma unroll
  for (int j = 0; j < D / 32; ++j) {
    int c = j * 32 + lane;
    float yv = y[c];
    v1 += yv * a[c];
    v2 += yv * a[D + c];
  }
  v1 = wave_sum(v1);
  v2 = wave_sum(v2);
  if (lane == 0) { s1[n] = v1; s2[n] = v2; }
}

// Per-node softmax over a dense row with (E - dv[n]) implicit zeros.
// One wave per node: DEG==32 lanes, pure shfl reductions.
// Outputs bg[n] (background prob) and coef[p] = attn_p - bg_n.
__global__ __launch_bounds__(256) void k_attn(const int* __restrict__ pe,
                                              const float* __restrict__ s1,
                                              const float* __restrict__ s2,
                                              const float* __restrict__ dv,
                                              float* __restrict__ bg,
                                              float* __restrict__ coef) {
  int n = blockIdx.x * 8 + (threadIdx.x >> 5);
  int lane = threadIdx.x & 31;
  if (n >= NN) return;
  int p = n * DEG + lane;
  float l = s1[n] + s2[pe[p]];
  l = (l > 0.0f) ? l : ALPHA * l;                 // leaky relu
  float m = fmaxf(wave_max(l), 0.0f);             // zeros participate
  float ex = __expf(l - m);
  float es = wave_sum(ex);
  float e0 = __expf(-m);                          // exp of the zero entries
  float Z  = es + ((float)NE - dv[n]) * e0;
  float b  = e0 / Z;
  coef[p]  = ex / Z - b;
  if (lane == 0) bg[n] = b;
}

// Bvec = sum_n bg[n] * X_proj[n]   (rank-1 softmax background)
#define BROWS 250
__global__ __launch_bounds__(256) void k_bvec(const float* __restrict__ bg,
                                              const float* __restrict__ Xp,
                                              float* __restrict__ Bvec) {
  int c = threadIdx.x;
  int r0 = blockIdx.x * BROWS;
  int r1 = (r0 + BROWS < NN) ? r0 + BROWS : NN;
  float acc = 0.0f;
  for (int r = r0; r < r1; ++r) acc += bg[r] * Xp[(size_t)r * D + c];
  atomicAdd(&Bvec[c], acc);
}

// EF[e] += coef[p] * X_proj[n]   (sparse correction of attn.T @ X_proj)
__global__ __launch_bounds__(256) void k_scatter_EF(const int* __restrict__ pn,
                                                    const int* __restrict__ pe,
                                                    const float* __restrict__ coef,
                                                    const float* __restrict__ Xp,
                                                    float* __restrict__ EF) {
  const int base = blockIdx.x * PPB;
  const int c = threadIdx.x;
  int ns[PPB], es[PPB];
  float cf[PPB];
  #pragma unroll
  for (int q = 0; q < PPB; ++q) {
    ns[q] = pn[base + q];
    es[q] = pe[base + q];
    cf[q] = coef[base + q];
  }
  #pragma unroll
  for (int q = 0; q < PPB; ++q) {
    float v = cf[q] * Xp[(size_t)ns[q] * D + c];
    atomicAdd(&EF[(size_t)es[q] * D + c], v);
  }
}

// out[n] = sum_j (EF[e_j] + Bvec) + bias = sum_j EF[e_j] + dv[n]*Bvec + bias
__global__ __launch_bounds__(256) void k_out(const int* __restrict__ pe,
                                             const float* __restrict__ EF,
                                             const float* __restrict__ dv,
                                             const float* __restrict__ Bvec,
                                             const float* __restrict__ bias,
                                             float* __restrict__ out) {
  int n = blockIdx.x, c = threadIdx.x;
  __shared__ int se[DEG];
  if (threadIdx.x < DEG) se[threadIdx.x] = pe[n * DEG + threadIdx.x];
  __syncthreads();
  float acc = 0.0f;
#ifdef HAVE_ASYNC_LDS
  __shared__ float tile[DEG * D];                 // 32KB staged gather
  #pragma unroll
  for (int it = 0; it < (DEG * D / 4) / 256; ++it) {
    int chunk = it * 256 + threadIdx.x;
    int row = chunk >> 6;
    int col = (chunk & 63) << 2;
    async_b128_to_lds(EF + (size_t)se[row] * D + col, &tile[row * D + col]);
  }
  __builtin_amdgcn_s_wait_asynccnt(0);
  __syncthreads();
  #pragma unroll 8
  for (int j = 0; j < DEG; ++j) acc += tile[j * D + c];
#else
  #pragma unroll 8
  for (int j = 0; j < DEG; ++j) acc += EF[(size_t)se[j] * D + c];
#endif
  out[n * D + c] = acc + dv[n] * Bvec[c] + bias[c];
}

// ------------------------------ launcher -----------------------------------

extern "C" void kernel_launch(void* const* d_in, const int* in_sizes, int n_in,
                              void* d_out, int out_size, void* d_ws, size_t ws_size,
                              hipStream_t stream) {
  const float* x      = (const float*)d_in[0];
  const float* weight = (const float*)d_in[1];
  const float* a      = (const float*)d_in[2];
  const float* bias   = (const float*)d_in[3];
  const int*   pn     = (const int*)d_in[4];
  const int*   pe     = (const int*)d_in[5];
  float*       out    = (float*)d_out;

  char* ws = (char*)d_ws;
  size_t off = 0;
  auto take = [&](size_t bytes) -> char* {
    char* p = ws + off;
    off = (off + bytes + 255) & ~(size_t)255;
    return p;
  };

  unsigned short* xb  = (unsigned short*)take((size_t)NN * D * 2);
  unsigned short* wsw = (unsigned short*)take((size_t)(D / 32) * (D / 16) * 32 * 16 * 2);
  float* Xp    = (float*)take((size_t)NN * D * 4);
  float* Yh    = (float*)take((size_t)NN * D * 4);
  float* s1    = (float*)take((size_t)NN * 4);
  float* s2    = (float*)take((size_t)NN * 4);
  float* bg    = (float*)take((size_t)NN * 4);
  float* coef  = (float*)take((size_t)NNZ * 4);
  float* dvinv = (float*)take((size_t)NN * 4);
  float* deinv = (float*)take((size_t)NE * 4);
  // contiguous accumulation region (zeroed every call -> graph-replay safe)
  const int accCount = NN + NE + 2 * NE * D + D;
  float* accbase = (float*)take((size_t)accCount * 4);
  float* dv   = accbase;
  float* de   = dv + NN;
  float* E2   = de + NE;
  float* EF   = E2 + (size_t)NE * D;
  float* Bvec = EF + (size_t)NE * D;

  const int T = 256;
  // 0) zero accumulators
  k_zero<<<(accCount + T - 1) / T, T, 0, stream>>>(accbase, accCount);
  // 1) bf16 conversions / B-fragment swizzle
  k_tobf<<<(NN * D + T - 1) / T, T, 0, stream>>>(x, xb, NN * D);
  k_swz<<<((D / 32) * (D / 16) * 32 + T - 1) / T, T, 0, stream>>>(weight, wsw);
  // 2) degrees
  k_count<<<(NNZ + T - 1) / T, T, 0, stream>>>(pn, pe, dv, de);
  k_invdeg<<<(NN + NE + T - 1) / T, T, 0, stream>>>(dv, de, dvinv, deinv);
  // 3) X_proj = x @ W  (WMMA bf16)
  {
    int waves = (NN / 16) * (D / 64);              // 7500 16x64 tiles
    k_gemm<<<(waves + 7) / 8, T, 0, stream>>>(xb, wsw, Xp);
  }
  // 4) E_feat scatter, then Y_hat gather
  k_scatter_E<<<NNZ / PPB, T, 0, stream>>>(pn, pe, Xp, dvinv, E2);
  k_yhat<<<NN, T, 0, stream>>>(pe, E2, deinv, dvinv, Xp, Yh);
  // 5) attention scores + per-node dense-row softmax (wave per node)
  k_scores<<<(NN + 7) / 8, T, 0, stream>>>(Yh, a, s1, s2);
  k_attn<<<(NN + 7) / 8, T, 0, stream>>>(pe, s1, s2, dv, bg, coef);
  // 6) attn.T @ X_proj = background rank-1 + sparse correction
  k_bvec<<<(NN + BROWS - 1) / BROWS, T, 0, stream>>>(bg, Xp, Bvec);
  k_scatter_EF<<<NNZ / PPB, T, 0, stream>>>(pn, pe, coef, Xp, EF);
  // 7) final gather back to nodes + bias
  k_out<<<NN, T, 0, stream>>>(pe, EF, dv, Bvec, bias, out);

  (void)in_sizes; (void)n_in; (void)out_size; (void)ws_size;
}
